// DualImageAttnProcessor_32315333935361
// MI455X (gfx1250) — matrix operand order repaired and checked by reading, not compiled, not run on hardware
//
#include <hip/hip_runtime.h>

// ---------------------------------------------------------------------------
// Types for CDNA5 WMMA (wave32): v_wmma_f32_16x16x32_bf16
// ---------------------------------------------------------------------------
typedef __bf16 bf16_t;
typedef __attribute__((ext_vector_type(16))) __bf16 v16bf;
typedef __attribute__((ext_vector_type(8)))  __bf16 v8bf;
typedef __attribute__((ext_vector_type(4)))  __bf16 v4bf;
typedef __attribute__((ext_vector_type(8)))  float  v8f;
typedef __attribute__((ext_vector_type(4)))  unsigned int v4u;
typedef __attribute__((ext_vector_type(8)))  int v8i;
typedef __attribute__((ext_vector_type(4)))  int v4i;

__device__ __forceinline__ v8f zero8() {
  v8f z = {0.f, 0.f, 0.f, 0.f, 0.f, 0.f, 0.f, 0.f};
  return z;
}

__device__ __forceinline__ v8f wmma_bf16(v16bf a, v16bf b, v8f c) {
  return __builtin_amdgcn_wmma_f32_16x16x32_bf16(false, a, false, b, (short)0, c,
                                                 false, false);
}

// LDS byte offset of a __shared__ object: low 32 bits of the generic pointer
// (shared aperture keeps the DS-space offset in addr[31:0]).
__device__ __forceinline__ unsigned lds_off(const void* p) {
  return (unsigned)(unsigned long long)p;
}

// ---------------------------------------------------------------------------
// Tensor Data Mover: 2D bf16 tile (tile_d1 rows x tile_d0 elems) from global
// (row stride = stride0 elems) into LDS at lds_addr, row-major contiguous.
// OOB rows/cols (vs tensor_d0/tensor_d1 remaining extents) are zero-filled.
// D# packing per CDNA5 ISA 8.3/8.4 (count=1, type=2(image), data_size=2B).
// 6-arg builtin on this toolchain: (v4u, v8i, v4i, v4i, v8i, cpol).
// ---------------------------------------------------------------------------
__device__ __forceinline__ void tdm_load_2d_bf16(
    unsigned lds_addr, const bf16_t* gaddr,
    unsigned tensor_d0, unsigned tensor_d1, unsigned stride0,
    unsigned tile_d0, unsigned tile_d1) {
  unsigned long long ga = (unsigned long long)gaddr;
  v4u g0;
  g0[0] = 1u;                                             // count=1
  g0[1] = lds_addr;                                       // lds_addr
  g0[2] = (unsigned)(ga & 0xffffffffu);                   // global_addr lo
  g0[3] = (unsigned)((ga >> 32) & 0x01ffffffu) | 0x80000000u;  // hi | type=2
  v8i g1;
  g1[0] = (int)(1u << 16);                                // data_size=1 (2B)
  g1[1] = (int)((tensor_d0 & 0xffffu) << 16);             // tensor_dim0 lo16
  g1[2] = (int)((tensor_d0 >> 16) | ((tensor_d1 & 0xffffu) << 16));
  g1[3] = (int)((tensor_d1 >> 16) | (tile_d0 << 16));     // tile_dim0
  g1[4] = (int)(tile_d1 & 0xffffu);                       // tile_dim1 (dim2=0)
  g1[5] = (int)stride0;                                   // dim0_stride lo32
  g1[6] = 0;                                              // stride hi / dim1_stride lo
  g1[7] = 0;
  v4i z4 = {0, 0, 0, 0};
  v8i z8 = {0, 0, 0, 0, 0, 0, 0, 0};
  __builtin_amdgcn_tensor_load_to_lds(g0, g1, z4, z4, z8, 0);
}

// A fragment: 16x32 bf16 tile, LDS row-major [m][k].
__device__ __forceinline__ v16bf load_frag_a(const bf16_t* base, int stride) {
  int lane = threadIdx.x & 31;
  int m  = lane & 15;
  int kb = (lane >> 4) << 3;
  const bf16_t* p = base + m * stride + kb;
  union { v16bf v; v8bf h[2]; } u;
  u.h[0] = *reinterpret_cast<const v8bf*>(p);
  u.h[1] = *reinterpret_cast<const v8bf*>(p + 16);
  return u.v;
}

// B fragment: 32x16 operand from LDS staged TRANSPOSED as [n][k].
__device__ __forceinline__ v16bf load_frag_b(const bf16_t* base, int stride) {
  int lane = threadIdx.x & 31;
  int n  = lane & 15;
  int ko = (lane >> 4) << 3;
  const bf16_t* p = base + n * stride + ko;
  union { v16bf v; v8bf h[2]; } u;
  u.h[0] = *reinterpret_cast<const v8bf*>(p);
  u.h[1] = *reinterpret_cast<const v8bf*>(p + 16);
  return u.v;
}

// ---------------------------------------------------------------------------
// Weight prep: W[K,N] f32 -> Wt[N,K] bf16 (one-time; B tiles become TDM-able)
// ---------------------------------------------------------------------------
__global__ __launch_bounds__(256) void transpose_cvt(const float* __restrict__ W,
                                                     bf16_t* __restrict__ Wt,
                                                     int K, int N) {
  __shared__ float t[32][33];
  const int k0 = blockIdx.x * 32, n0 = blockIdx.y * 32;
  const int tx = threadIdx.x & 31, ty = threadIdx.x >> 5;
#pragma unroll
  for (int r = ty; r < 32; r += 8) t[r][tx] = W[(size_t)(k0 + r) * N + n0 + tx];
  __syncthreads();
#pragma unroll
  for (int r = ty; r < 32; r += 8)
    Wt[(size_t)(n0 + r) * K + k0 + tx] = (bf16_t)t[tx][r];
}

// ---------------------------------------------------------------------------
// GEMM: C[M,N] = A[M,K] * Wt[N,K]^T + biasScale*bias.  Tile 128x128x32.
// Double-buffered LDS: stage i+1 (TDM and/or f32 loads) overlaps stage-i WMMA.
// TDM completion is in-order per wave, so waiting tensorcnt<=n_inflight_next
// guarantees the current stage is resident.
// ---------------------------------------------------------------------------
#define GT_M 128
#define GT_N 128
#define GT_K 32

template <bool A_BF16, bool OUT_BF16>
__global__ __launch_bounds__(256) void gemm_wmma(
    const void* __restrict__ Aptr, const bf16_t* __restrict__ Wt,
    const float* __restrict__ bias, float biasScale,
    void* __restrict__ Cptr, int M, int K, int N) {
  __shared__ __align__(16) bf16_t As[2][GT_M][GT_K];
  __shared__ __align__(16) bf16_t Bs[2][GT_N][GT_K];

  const int tid  = threadIdx.x;
  const int lane = tid & 31;
  const int wave = tid >> 5;
  const int wr = wave & 3;
  const int wc = wave >> 2;
  const int m0 = blockIdx.x * GT_M;
  const int n0 = blockIdx.y * GT_N;
  const int nk = K / GT_K;

  v8f acc[2][4];
#pragma unroll
  for (int i = 0; i < 2; ++i)
#pragma unroll
    for (int j = 0; j < 4; ++j) acc[i][j] = zero8();

  const int arow  = tid >> 1;
  const int akseg = (tid & 1) * 16;

  auto issue_stage = [&](int kt, int buf) {
    // B tile: 128 n-rows x 32 k, stride K (pre-transposed bf16 weights)
    tdm_load_2d_bf16(lds_off(&Bs[buf][0][0]), Wt + (size_t)n0 * K + kt,
                     (unsigned)(K - kt), (unsigned)(N - n0), (unsigned)K,
                     GT_K, GT_N);
    if (A_BF16) {
      tdm_load_2d_bf16(lds_off(&As[buf][0][0]),
                       (const bf16_t*)Aptr + (size_t)m0 * K + kt,
                       (unsigned)(K - kt), (unsigned)(M - m0), (unsigned)K,
                       GT_K, GT_M);
    }
  };

  auto stage_manual = [&](int kt, int buf) {
    const float* Af = (const float*)Aptr;
    const int gr = m0 + arow;
    if (gr < M) {
      const float* ap = Af + (size_t)gr * K + kt + akseg;
      if (kt + GT_K < K) __builtin_prefetch(ap + GT_K, 0, 0);
#pragma unroll
      for (int v = 0; v < 16; v += 4) {
        float4 f = *reinterpret_cast<const float4*>(ap + v);
        v4bf t = {(bf16_t)f.x, (bf16_t)f.y, (bf16_t)f.z, (bf16_t)f.w};
        *reinterpret_cast<v4bf*>(&As[buf][arow][akseg + v]) = t;
      }
    } else {
      v4bf z = {(bf16_t)0.f, (bf16_t)0.f, (bf16_t)0.f, (bf16_t)0.f};
#pragma unroll
      for (int v = 0; v < 16; v += 4)
        *reinterpret_cast<v4bf*>(&As[buf][arow][akseg + v]) = z;
    }
  };

  // ---- prologue: stage 0 into buffer 0 ----
  if (wave == 0) issue_stage(0, 0);
  if (!A_BF16) stage_manual(0, 0);

  for (int it = 0; it < nk; ++it) {
    const int cur = it & 1;
    const int nxt = cur ^ 1;
    const bool has_next = (it + 1) < nk;

    __syncthreads();  // readers of buf[nxt] (iteration it-1) are done
    if (has_next) {
      if (wave == 0) issue_stage((it + 1) * GT_K, nxt);
      if (!A_BF16) stage_manual((it + 1) * GT_K, nxt);
    }
    if (wave == 0) {
      if (has_next)
        __builtin_amdgcn_s_wait_tensorcnt(A_BF16 ? 2 : 1);  // stage it done
      else
        __builtin_amdgcn_s_wait_tensorcnt(0);
    }
    __syncthreads();  // publish buf[cur] to all waves

    v16bf af[2];
    af[0] = load_frag_a(&As[cur][wr * 32][0], GT_K);
    af[1] = load_frag_a(&As[cur][wr * 32 + 16][0], GT_K);
    v16bf bf_[4];
#pragma unroll
    for (int ni = 0; ni < 4; ++ni)
      bf_[ni] = load_frag_b(&Bs[cur][wc * 64 + ni * 16][0], GT_K);
#pragma unroll
    for (int mi = 0; mi < 2; ++mi)
#pragma unroll
      for (int ni = 0; ni < 4; ++ni)
        acc[mi][ni] = wmma_bf16(af[mi], bf_[ni], acc[mi][ni]);
  }

  const int ncol   = lane & 15;
  const int rowoff = (lane >> 4) * 8;
#pragma unroll
  for (int mi = 0; mi < 2; ++mi) {
#pragma unroll
    for (int ni = 0; ni < 4; ++ni) {
      const int col = n0 + wc * 64 + ni * 16 + ncol;
      const float bv = bias ? biasScale * bias[col] : 0.f;
#pragma unroll
      for (int j = 0; j < 8; ++j) {
        const int row = m0 + wr * 32 + mi * 16 + rowoff + j;
        if (row < M) {
          const float v = acc[mi][ni][j] + bv;
          if (OUT_BF16)
            ((bf16_t*)Cptr)[(size_t)row * N + col] = (bf16_t)v;
          else
            ((float*)Cptr)[(size_t)row * N + col] = v;
        }
      }
    }
  }
}

// ---------------------------------------------------------------------------
// Attention (all operands bf16): per (128 q-rows, b, h); 3 branches summed.
// ---------------------------------------------------------------------------
#define AC   1280
#define AH   20
#define ADH  64
#define ASQ  4096
#define ASCALE 0.125f

template <int STILES, int PVK>
__device__ __forceinline__ void attn_branch(
    const bf16_t* __restrict__ Kp, const bf16_t* __restrict__ Vp, int Sk,
    int b, int h, int wave, int lane, int tid, int m0,
    bf16_t (&Qs)[128][64], bf16_t (&Ks)[80][64], bf16_t (&VTs)[64][96],
    bf16_t (&Ps)[8][16][96], v8f (&accO)[4]) {
  const int pvpad = PVK * 32;  // 96 or 32

  __syncthreads();  // previous branch done reading Ks/VTs
  // ---- K rows [key][d] via TDM; rows >= Sk zero-filled by OOB handling ----
  if (wave == 0) {
    tdm_load_2d_bf16(lds_off(&Ks[0][0]),
                     Kp + (size_t)(b * Sk) * AC + h * ADH,
                     (unsigned)(AC - h * ADH), (unsigned)Sk, AC,
                     ADH, STILES * 16);
  }
  // ---- V transposed [d][key] staged manually (TDM cannot transpose) ----
  {
    const int key = tid >> 1, dseg = (tid & 1) * 32;
    if (key < pvpad) {
      if (key < Sk) {
        const v8bf* vp =
            (const v8bf*)(Vp + (size_t)(b * Sk + key) * AC + h * ADH + dseg);
#pragma unroll
        for (int c = 0; c < 4; ++c) {
          v8bf x = vp[c];
#pragma unroll
          for (int e = 0; e < 8; ++e) VTs[dseg + c * 8 + e][key] = x[e];
        }
      } else {
#pragma unroll
        for (int v = 0; v < 32; ++v) VTs[dseg + v][key] = (bf16_t)0.f;
      }
    }
  }
  if (wave == 0) __builtin_amdgcn_s_wait_tensorcnt(0);  // K (and Q) resident
  __syncthreads();

  // ---- zero this wave's P scratch (covers PV K-padding) ----
  {
    unsigned int* pz = reinterpret_cast<unsigned int*>(&Ps[wave][0][0]);
#pragma unroll
    for (int idx = 0; idx < 24; ++idx) pz[lane + 32 * idx] = 0u;
  }

  // ---- scores: S = Q * K^T ----
  v8f st[STILES];
#pragma unroll
  for (int nt = 0; nt < STILES; ++nt) {
    v8f s = zero8();
#pragma unroll
    for (int ks = 0; ks < 2; ++ks) {
      v16bf aq = load_frag_a(&Qs[m0][ks * 32], 64);
      v16bf bk_ = load_frag_b(&Ks[nt * 16][ks * 32], 64);
      s = wmma_bf16(aq, bk_, s);
    }
    st[nt] = s;
  }

  // ---- masked softmax per row ----
  const int ncol = lane & 15;
#pragma unroll
  for (int j = 0; j < 8; ++j) {
    float mx = -1e30f;
#pragma unroll
    for (int nt = 0; nt < STILES; ++nt) {
      const int key = nt * 16 + ncol;
      float s = st[nt][j] * ASCALE;
      s = (key < Sk) ? s : -1e30f;
      st[nt][j] = s;
      mx = fmaxf(mx, s);
    }
#pragma unroll
    for (int off = 1; off < 16; off <<= 1) mx = fmaxf(mx, __shfl_xor(mx, off, 16));
    float sum = 0.f;
#pragma unroll
    for (int nt = 0; nt < STILES; ++nt) {
      float p = __expf(st[nt][j] - mx);
      st[nt][j] = p;
      sum += p;
    }
#pragma unroll
    for (int off = 1; off < 16; off <<= 1) sum += __shfl_xor(sum, off, 16);
    const float inv = 1.f / sum;
#pragma unroll
    for (int nt = 0; nt < STILES; ++nt) st[nt][j] *= inv;
  }

  // ---- store P into this wave's LDS scratch (A-operand source) ----
  const int prow = (lane >> 4) * 8;
#pragma unroll
  for (int j = 0; j < 8; ++j) {
    const int m = prow + j;
#pragma unroll
    for (int nt = 0; nt < STILES; ++nt)
      Ps[wave][m][nt * 16 + ncol] = (bf16_t)st[nt][j];
  }

  // ---- O += P * V ----
  v16bf ap[PVK];
#pragma unroll
  for (int ks = 0; ks < PVK; ++ks)
    ap[ks] = load_frag_a(&Ps[wave][0][ks * 32], 96);
#pragma unroll
  for (int dt = 0; dt < 4; ++dt) {
    v8f acc = accO[dt];
#pragma unroll
    for (int ks = 0; ks < PVK; ++ks) {
      v16bf bv = load_frag_b(&VTs[dt * 16][ks * 32], 96);
      acc = wmma_bf16(ap[ks], bv, acc);
    }
    accO[dt] = acc;
  }
}

__global__ __launch_bounds__(256) void attn_wmma(
    const bf16_t* __restrict__ Q,
    const bf16_t* __restrict__ Kt, const bf16_t* __restrict__ Vt,
    const bf16_t* __restrict__ Kid, const bf16_t* __restrict__ Vid,
    const bf16_t* __restrict__ Kh, const bf16_t* __restrict__ Vh,
    bf16_t* __restrict__ O) {
  const int qb = blockIdx.x;
  const int bh = blockIdx.y;
  const int b = bh / AH, h = bh % AH;
  const int tid = threadIdx.x, lane = tid & 31, wave = tid >> 5;
  const int m0 = wave * 16;

  __shared__ __align__(16) bf16_t Qs[128][64];
  __shared__ __align__(16) bf16_t Ks[80][64];
  __shared__ __align__(16) bf16_t VTs[64][96];
  __shared__ __align__(16) bf16_t Ps[8][16][96];

  // ---- Q block via TDM (waited inside first branch, same-wave in-order) ----
  if (wave == 0) {
    tdm_load_2d_bf16(lds_off(&Qs[0][0]),
                     Q + (size_t)(b * ASQ + qb * 128) * AC + h * ADH,
                     (unsigned)(AC - h * ADH),
                     (unsigned)(ASQ - qb * 128), AC, ADH, 128);
  }

  v8f accO[4];
#pragma unroll
  for (int i = 0; i < 4; ++i) accO[i] = zero8();

  attn_branch<5, 3>(Kt, Vt, 77, b, h, wave, lane, tid, m0, Qs, Ks, VTs, Ps, accO);
  attn_branch<1, 1>(Kid, Vid, 16, b, h, wave, lane, tid, m0, Qs, Ks, VTs, Ps, accO);
  attn_branch<1, 1>(Kh, Vh, 16, b, h, wave, lane, tid, m0, Qs, Ks, VTs, Ps, accO);

  const int ncol = lane & 15;
  const int rowoff = (lane >> 4) * 8;
#pragma unroll
  for (int dt = 0; dt < 4; ++dt) {
#pragma unroll
    for (int j = 0; j < 8; ++j) {
      const int row = qb * 128 + m0 + rowoff + j;
      O[((size_t)(b * ASQ + row)) * AC + h * ADH + dt * 16 + ncol] =
          (bf16_t)accO[dt][j];
    }
  }
}

// ---------------------------------------------------------------------------
// Host launch
// ---------------------------------------------------------------------------
extern "C" void kernel_launch(void* const* d_in, const int* in_sizes, int n_in,
                              void* d_out, int out_size, void* d_ws, size_t ws_size,
                              hipStream_t stream) {
  const float* hidden  = (const float*)d_in[0];
  const float* text    = (const float*)d_in[1];
  const float* ids     = (const float*)d_in[2];
  const float* hair    = (const float*)d_in[3];
  const float* Wq      = (const float*)d_in[4];
  const float* Wk      = (const float*)d_in[5];
  const float* Wv      = (const float*)d_in[6];
  const float* Wo      = (const float*)d_in[7];
  const float* b_o     = (const float*)d_in[8];
  const float* Wk_id   = (const float*)d_in[9];
  const float* Wv_id   = (const float*)d_in[10];
  const float* Wk_hair = (const float*)d_in[11];
  const float* Wv_hair = (const float*)d_in[12];
  float* out = (float*)d_out;

  const int B = 8, SQ = 4096, ST = 77, SI = 16, C = 1280, DC = 2048;
  const int MQ = B * SQ, MT = B * ST, MI = B * SI;

  char* w = (char*)d_ws;
  bf16_t* wq_t  = (bf16_t*)w; w += (size_t)C * C * 2;
  bf16_t* wo_t  = (bf16_t*)w; w += (size_t)C * C * 2;
  bf16_t* wk_t  = (bf16_t*)w; w += (size_t)DC * C * 2;
  bf16_t* wv_t  = (bf16_t*)w; w += (size_t)DC * C * 2;
  bf16_t* wki_t = (bf16_t*)w; w += (size_t)DC * C * 2;
  bf16_t* wvi_t = (bf16_t*)w; w += (size_t)DC * C * 2;
  bf16_t* wkh_t = (bf16_t*)w; w += (size_t)DC * C * 2;
  bf16_t* wvh_t = (bf16_t*)w; w += (size_t)DC * C * 2;
  bf16_t* q_buf   = (bf16_t*)w; w += (size_t)MQ * C * 2;
  bf16_t* attnsum = (bf16_t*)w; w += (size_t)MQ * C * 2;
  bf16_t* k_text  = (bf16_t*)w; w += (size_t)MT * C * 2;
  bf16_t* v_text  = (bf16_t*)w; w += (size_t)MT * C * 2;
  bf16_t* k_id    = (bf16_t*)w; w += (size_t)MI * C * 2;
  bf16_t* v_id    = (bf16_t*)w; w += (size_t)MI * C * 2;
  bf16_t* k_hair  = (bf16_t*)w; w += (size_t)MI * C * 2;
  bf16_t* v_hair  = (bf16_t*)w; w += (size_t)MI * C * 2;

  dim3 blk(256);
  const int NB = C / GT_N;  // 10

  // ---- one-time weight transpose+convert: W[K,N]f32 -> Wt[N,K]bf16 ----
  transpose_cvt<<<dim3(C / 32, C / 32), blk, 0, stream>>>(Wq, wq_t, C, C);
  transpose_cvt<<<dim3(C / 32, C / 32), blk, 0, stream>>>(Wo, wo_t, C, C);
  transpose_cvt<<<dim3(DC / 32, C / 32), blk, 0, stream>>>(Wk, wk_t, DC, C);
  transpose_cvt<<<dim3(DC / 32, C / 32), blk, 0, stream>>>(Wv, wv_t, DC, C);
  transpose_cvt<<<dim3(DC / 32, C / 32), blk, 0, stream>>>(Wk_id, wki_t, DC, C);
  transpose_cvt<<<dim3(DC / 32, C / 32), blk, 0, stream>>>(Wv_id, wvi_t, DC, C);
  transpose_cvt<<<dim3(DC / 32, C / 32), blk, 0, stream>>>(Wk_hair, wkh_t, DC, C);
  transpose_cvt<<<dim3(DC / 32, C / 32), blk, 0, stream>>>(Wv_hair, wvh_t, DC, C);

  // ---- projections (A=f32 external input, out=bf16) ----
  gemm_wmma<false, true><<<dim3((MQ + GT_M - 1) / GT_M, NB), blk, 0, stream>>>(
      hidden, wq_t, nullptr, 0.f, q_buf, MQ, C, C);
  gemm_wmma<false, true><<<dim3((MT + GT_M - 1) / GT_M, NB), blk, 0, stream>>>(
      text, wk_t, nullptr, 0.f, k_text, MT, DC, C);
  gemm_wmma<false, true><<<dim3((MT + GT_M - 1) / GT_M, NB), blk, 0, stream>>>(
      text, wv_t, nullptr, 0.f, v_text, MT, DC, C);
  gemm_wmma<false, true><<<dim3((MI + GT_M - 1) / GT_M, NB), blk, 0, stream>>>(
      ids, wki_t, nullptr, 0.f, k_id, MI, DC, C);
  gemm_wmma<false, true><<<dim3((MI + GT_M - 1) / GT_M, NB), blk, 0, stream>>>(
      ids, wvi_t, nullptr, 0.f, v_id, MI, DC, C);
  gemm_wmma<false, true><<<dim3((MI + GT_M - 1) / GT_M, NB), blk, 0, stream>>>(
      hair, wkh_t, nullptr, 0.f, k_hair, MI, DC, C);
  gemm_wmma<false, true><<<dim3((MI + GT_M - 1) / GT_M, NB), blk, 0, stream>>>(
      hair, wvh_t, nullptr, 0.f, v_hair, MI, DC, C);

  // ---- fused 3-branch attention, outputs summed (Wo is linear) ----
  attn_wmma<<<dim3(SQ / 128, B * AH), blk, 0, stream>>>(
      q_buf, k_text, v_text, k_id, v_id, k_hair, v_hair, attnsum);

  // ---- single output projection (A=bf16 TDM path): sum @ Wo + 3*b_o ----
  gemm_wmma<true, false><<<dim3((MQ + GT_M - 1) / GT_M, NB), blk, 0, stream>>>(
      attnsum, wo_t, b_o, 3.0f, out, MQ, C, C);
}
